// SingleHeadAttentionWrapper_2911987827479
// MI455X (gfx1250) — matrix operand order repaired and checked by reading, not compiled, not run on hardware
//
#include <hip/hip_runtime.h>

typedef __attribute__((ext_vector_type(16))) _Float16 v16h;
typedef __attribute__((ext_vector_type(8)))  _Float16 v8h;
typedef __attribute__((ext_vector_type(4)))  _Float16 v4h;
typedef __attribute__((ext_vector_type(8)))  float    v8f;
typedef __attribute__((ext_vector_type(4)))  float    f32x4;

#define S_LEN 4096
#define D_DIM 128
#define KSTR  136   // halves per K row in LDS (pad to dodge bank conflicts)
#define VSTR  36    // halves per V^T row in LDS
#define PSTR  40    // halves per P row in LDS scratch

union V16 { v16h v; v8h h8[2]; v4h h4[4]; };

// Max-reduce across each 16-lane DPP row (lanes 0-15 and 16-31 independently).
// Pure VALU: no DS traffic, no counter waits. Requires EXEC all-ones (true here).
__device__ __forceinline__ float dpp_rowmax16(float x) {
    float t;
    t = __uint_as_float((unsigned)__builtin_amdgcn_mov_dpp(
            (int)__float_as_uint(x), 0xB1, 0xF, 0xF, true)); // quad_perm [1,0,3,2]
    x = fmaxf(x, t);
    t = __uint_as_float((unsigned)__builtin_amdgcn_mov_dpp(
            (int)__float_as_uint(x), 0x4E, 0xF, 0xF, true)); // quad_perm [2,3,0,1]
    x = fmaxf(x, t);
    t = __uint_as_float((unsigned)__builtin_amdgcn_mov_dpp(
            (int)__float_as_uint(x), 0x141, 0xF, 0xF, true)); // row_half_mirror
    x = fmaxf(x, t);
    t = __uint_as_float((unsigned)__builtin_amdgcn_mov_dpp(
            (int)__float_as_uint(x), 0x140, 0xF, 0xF, true)); // row_mirror
    x = fmaxf(x, t);
    return x;
}

__global__ __launch_bounds__(128)
void fa_causal_f16wmma(const float* __restrict__ Q,
                       const float* __restrict__ K,
                       const float* __restrict__ V,
                       float* __restrict__ O)
{
    __shared__ __align__(16) _Float16 sK[32 * KSTR];     // K tile, row-major [key][d]
    __shared__ __align__(16) _Float16 sV[D_DIM * VSTR];  // V^T tile [d][key]
    __shared__ __align__(16) _Float16 sP[4 * 16 * PSTR]; // per-wave P scratch

    const int tid  = threadIdx.x;
    const int wave = tid >> 5;
    const int lane = tid & 31;
    const int ln   = lane & 15;   // N / row-in-group index
    const int hi   = lane >> 4;   // lane-group (0: lanes 0-15, 1: lanes 16-31)

    const int b  = blockIdx.x >> 6;          // batch index (S/64 == 64 blocks per batch)
    const int qb = (blockIdx.x & 63) * 64;   // block's first query row
    const int qw = qb + wave * 16;           // wave's first query row

    const float scale = 0.08838834764831845f; // 1/sqrt(128), folded into Q
    const float NEGINF = -__builtin_inff();

    // ---- Load this wave's 16x128 Q tile (pre-scaled) into WMMA A-layout regs ----
    // A-layout (16-bit, 16x32): element j of lane L -> K = j + (j>=8?8:0) + (L>=16?8:0)
    const float* qrow = Q + ((size_t)b * S_LEN + qw + ln) * D_DIM;
    V16 qa[4];
#pragma unroll
    for (int s = 0; s < 4; ++s) {
        int base = s * 32 + hi * 8;
        f32x4 f0 = *(const f32x4*)(qrow + base);
        f32x4 f1 = *(const f32x4*)(qrow + base + 4);
        f32x4 f2 = *(const f32x4*)(qrow + base + 16);
        f32x4 f3 = *(const f32x4*)(qrow + base + 20);
        v16h t;
#pragma unroll
        for (int j = 0; j < 4; ++j) {
            t[j]      = (_Float16)(f0[j] * scale);
            t[4 + j]  = (_Float16)(f1[j] * scale);
            t[8 + j]  = (_Float16)(f2[j] * scale);
            t[12 + j] = (_Float16)(f3[j] * scale);
        }
        qa[s].v = t;
    }

    // all-ones B tile (registers only): l-accumulator = P * ones via WMMA
    v16h ones;
#pragma unroll
    for (int j = 0; j < 16; ++j) ones[j] = (_Float16)1.0f;

    // ---- Running softmax state (C-layout rows: VGPR e -> row e + hi*8) ----
    v8f acc[8];
    v8f accl;           // running row-sum of P (denominator), via ones-tile WMMA
    float m_i[8];
#pragma unroll
    for (int e = 0; e < 8; ++e) { m_i[e] = NEGINF; accl[e] = 0.0f; }
#pragma unroll
    for (int t = 0; t < 8; ++t)
#pragma unroll
        for (int e = 0; e < 8; ++e) acc[t][e] = 0.0f;

    const float* Kg = K + (size_t)b * S_LEN * D_DIM;
    const float* Vg = V + (size_t)b * S_LEN * D_DIM;
    const int kv_end = qb + 64;

    for (int kv = 0; kv < kv_end; kv += 32) {
        __syncthreads(); // previous tile fully consumed
        // ---- cooperative load of 32x128 K and V tiles (f32 -> f16) ----
#pragma unroll
        for (int i = 0; i < 8; ++i) {
            int e4 = i * 512 + tid * 4;
            int kr = e4 >> 7;        // key row 0..31
            int dc = e4 & 127;       // d col, multiple of 4
            f32x4 kf = *(const f32x4*)(Kg + (size_t)(kv + kr) * D_DIM + dc);
            f32x4 vf = *(const f32x4*)(Vg + (size_t)(kv + kr) * D_DIM + dc);
            v4h kh;
#pragma unroll
            for (int j = 0; j < 4; ++j) kh[j] = (_Float16)kf[j];
            *(v4h*)&sK[kr * KSTR + dc] = kh;
#pragma unroll
            for (int j = 0; j < 4; ++j) sV[(dc + j) * VSTR + kr] = (_Float16)vf[j];
        }
        __syncthreads();

        if (kv <= qw + 15) {   // uniform per-wave: EXEC stays all-ones
            // ---- scores S = (Q*scale) K^T : 2 key-tiles x 4 K-steps of WMMA ----
            v8f s0, s1;
#pragma unroll
            for (int e = 0; e < 8; ++e) { s0[e] = 0.0f; s1[e] = 0.0f; }
#pragma unroll
            for (int s = 0; s < 4; ++s) {
                int col = s * 32 + hi * 8;   // B-layout: K = hi*8 + j (+8 for j>=8) within step
                V16 b0, b1;
                b0.h8[0] = *(const v8h*)&sK[ln * KSTR + col];
                b0.h8[1] = *(const v8h*)&sK[ln * KSTR + col + 16];
                b1.h8[0] = *(const v8h*)&sK[(16 + ln) * KSTR + col];
                b1.h8[1] = *(const v8h*)&sK[(16 + ln) * KSTR + col + 16];
                s0 = __builtin_amdgcn_wmma_f32_16x16x32_f16(false, qa[s].v, false, b0.v,
                                                            (short)0, s0, false, false);
                s1 = __builtin_amdgcn_wmma_f32_16x16x32_f16(false, qa[s].v, false, b1.v,
                                                            (short)0, s1, false, false);
            }

            // ---- causal mask + online max update (f32, DPP reductions) ----
            int key0 = kv + ln;
            int key1 = kv + 16 + ln;
            float ar[8];
#pragma unroll
            for (int e = 0; e < 8; ++e) {
                int row = qw + e + hi * 8;
                float a0 = (key0 <= row) ? s0[e] : NEGINF;
                float a1 = (key1 <= row) ? s1[e] : NEGINF;
                float rm = dpp_rowmax16(fmaxf(a0, a1));
                float mn = fmaxf(m_i[e], rm);
                float al = __expf(m_i[e] - mn);
                s0[e] = __expf(a0 - mn);
                s1[e] = __expf(a1 - mn);
                m_i[e] = mn;
                ar[e]  = al;
            }

            // rescale accumulators (output + denominator) by per-row alpha
#pragma unroll
            for (int t = 0; t < 8; ++t)
#pragma unroll
                for (int e = 0; e < 8; ++e) acc[t][e] *= ar[e];
#pragma unroll
            for (int e = 0; e < 8; ++e) accl[e] *= ar[e];

            // ---- P (C-layout) -> LDS -> A-layout for P*V ----
            int pbase = wave * 16 * PSTR;
#pragma unroll
            for (int e = 0; e < 8; ++e) {
                int row = e + hi * 8;
                sP[pbase + row * PSTR + ln]      = (_Float16)s0[e];
                sP[pbase + row * PSTR + 16 + ln] = (_Float16)s1[e];
            }
            asm volatile("s_wait_dscnt 0" ::: "memory"); // wave-private LDS RAW
            V16 pa;
            {
                int koff = hi * 8;
                const _Float16* pr = &sP[pbase + ln * PSTR + koff];
                pa.h4[0] = *(const v4h*)(pr);
                pa.h4[1] = *(const v4h*)(pr + 4);
                pa.h4[2] = *(const v4h*)(pr + 16);
                pa.h4[3] = *(const v4h*)(pr + 20);
            }

            // denominator: l += P * ones (one WMMA, register-constant B)
            accl = __builtin_amdgcn_wmma_f32_16x16x32_f16(false, pa.v, false, ones,
                                                          (short)0, accl, false, false);

            // ---- O += P * V : 8 d-tiles of WMMA against V^T from LDS ----
#pragma unroll
            for (int t = 0; t < 8; ++t) {
                const _Float16* vr = &sV[(t * 16 + ln) * VSTR + hi * 8];
                V16 bv;
                bv.h4[0] = *(const v4h*)(vr);
                bv.h4[1] = *(const v4h*)(vr + 4);
                bv.h4[2] = *(const v4h*)(vr + 16);
                bv.h4[3] = *(const v4h*)(vr + 20);
                acc[t] = __builtin_amdgcn_wmma_f32_16x16x32_f16(false, pa.v, false, bv.v,
                                                                (short)0, acc[t], false, false);
            }
        }
    }

    // ---- normalize and store (fp32) ----
    float* orow = O + ((size_t)b * S_LEN + qw) * D_DIM;
#pragma unroll
    for (int e = 0; e < 8; ++e) {
        float inv = 1.0f / accl[e];   // every column of accl holds the row sum
        int row = e + hi * 8;
#pragma unroll
        for (int t = 0; t < 8; ++t)
            orow[(size_t)row * D_DIM + t * 16 + ln] = acc[t][e] * inv;
    }
}

extern "C" void kernel_launch(void* const* d_in, const int* in_sizes, int n_in,
                              void* d_out, int out_size, void* d_ws, size_t ws_size,
                              hipStream_t stream) {
    (void)in_sizes; (void)n_in; (void)out_size; (void)d_ws; (void)ws_size;
    const float* q = (const float*)d_in[0];
    const float* k = (const float*)d_in[1];
    const float* v = (const float*)d_in[2];
    float* o = (float*)d_out;
    dim3 grid(4 * (S_LEN / 64));   // B * (S/64) = 256 blocks
    dim3 block(128);               // 4 waves; each wave owns a 16-row query tile
    fa_causal_f16wmma<<<grid, block, 0, stream>>>(q, k, v, o);
}
